// CustomSelfAttention_35742717837551
// MI455X (gfx1250) — matrix operand order repaired and checked by reading, not compiled, not run on hardware
//
#include <hip/hip_runtime.h>

typedef __attribute__((ext_vector_type(8)))  float  v8f;
typedef __attribute__((ext_vector_type(16))) __bf16 v16bf;
typedef __attribute__((ext_vector_type(2)))  __bf16 v2bf;
typedef int v4i __attribute__((__vector_size__(16)));

#define T_SEQ 2048
#define D_EMB 1024
#define N_HEADS 16
#define HD 64

// ---- f32 -> bf16 conversion helpers (prefer native packed cvt) -------------
__device__ __forceinline__ unsigned short f2bf(float f) {
  unsigned u = __float_as_uint(f);
  u += 0x7FFFu + ((u >> 16) & 1u);   // round-to-nearest-even
  return (unsigned short)(u >> 16);
}

__device__ __forceinline__ unsigned cvt2bf(float a, float b) {
#if __has_builtin(__builtin_amdgcn_cvt_pk_bf16_f32)
  v2bf t = __builtin_amdgcn_cvt_pk_bf16_f32(a, b);
  return __builtin_bit_cast(unsigned, t);
#else
  return (unsigned)f2bf(a) | ((unsigned)f2bf(b) << 16);
#endif
}

// ---- async global->LDS copy (gfx1250 ASYNCcnt path), guarded ---------------
#if __has_builtin(__builtin_amdgcn_global_load_async_to_lds_b128)
#define HAVE_ASYNC_LDS 1
#endif

__device__ __forceinline__ void wait_async_cnt0() {
#if defined(HAVE_ASYNC_LDS)
#if __has_builtin(__builtin_amdgcn_s_wait_asynccnt)
  __builtin_amdgcn_s_wait_asynccnt(0);
#else
  asm volatile("s_wait_asynccnt 0" ::: "memory");
#endif
#endif
}

union FragU { uint4 q[2]; v16bf v; };

// A-fragment (16x32 bf16). Row-major source, stride in halves (rows 16B aligned).
// lanes 0-15: K = kbase+{0..7, 16..23}; lanes 16-31: K = kbase+{8..15, 24..31}.
__device__ __forceinline__ v16bf load_a_frag(const unsigned short* p, int row,
                                             int stride, int kbase, int lane) {
  FragU u;
  int k0 = kbase + ((lane & 16) ? 8 : 0);
  const unsigned short* r = p + row * stride;
  u.q[0] = *(const uint4*)(r + k0);
  u.q[1] = *(const uint4*)(r + k0 + 16);
  return u.v;
}

// B-fragment (32x16 bf16) from N-major storage BT[n][k].
// lanes 0-15: K = kbase+{0..7, 16..23}; lanes 16-31: K = kbase+{8..15, 24..31}.
__device__ __forceinline__ v16bf load_b_frag(const unsigned short* p, int nrow,
                                             int stride, int kbase, int lane) {
  FragU u;
  int k0 = kbase + ((lane & 16) ? 8 : 0);
  const unsigned short* r = p + nrow * stride;
  u.q[0] = *(const uint4*)(r + k0);
  u.q[1] = *(const uint4*)(r + k0 + 16);
  return u.v;
}

// ---------------------------------------------------------------------------
// Kernel 1: QKV projection. grid = (N/128, M/128, 3), block = 256 (8 waves).
// Block tile 128x128, wave tile 32x64 (2x4 WMMA frags) -> 8 wmma / 12 ds_load.
// Outputs bf16 in [B, H, T, Hd] layout.
// ---------------------------------------------------------------------------
__global__ __launch_bounds__(256) void qkv_gemm_kernel(
    const float* __restrict__ x,
    const float* __restrict__ wq, const float* __restrict__ bq,
    const float* __restrict__ wk, const float* __restrict__ bk,
    const float* __restrict__ wv, const float* __restrict__ bv,
    unsigned short* __restrict__ Qb, unsigned short* __restrict__ Kb,
    unsigned short* __restrict__ Vb)
{
  __shared__ unsigned short At[128 * 40];  // A tile 128x32 bf16, stride 40 halves
  __shared__ unsigned short Bt[128 * 40];  // W tile N-major: Bt[n][k]

  const float* W; const float* bias; unsigned short* out;
  if (blockIdx.z == 0)      { W = wq; bias = bq; out = Qb; }
  else if (blockIdx.z == 1) { W = wk; bias = bk; out = Kb; }
  else                      { W = wv; bias = bv; out = Vb; }

  int tid = threadIdx.x, lane = tid & 31, wave = tid >> 5;
  int wm = wave >> 1, wn = wave & 1;           // 4x2 wave grid, 32x64 per wave
  int m0 = blockIdx.y * 128, n0 = blockIdx.x * 128;

  v8f acc[2][4];
  for (int mi = 0; mi < 2; ++mi)
    for (int ni = 0; ni < 4; ++ni)
      for (int v = 0; v < 8; ++v) acc[mi][ni][v] = 0.f;

  for (int kt = 0; kt < D_EMB; kt += 32) {
    // A: 128x32 fp32 -> bf16 LDS (1024 float4 slots, 4/thread)
    for (int i = tid; i < 1024; i += 256) {
      int r = i >> 3, c4 = i & 7;
      float4 d = *(const float4*)(x + (size_t)(m0 + r) * D_EMB + kt + c4 * 4);
      uint2 t;
      t.x = cvt2bf(d.x, d.y);
      t.y = cvt2bf(d.z, d.w);
      *(uint2*)(&At[r * 40 + c4 * 4]) = t;
    }
    // B: 32x128 fp32 -> bf16 transposed into Bt[n][k] (1024 float4, 4/thread)
    for (int i = tid; i < 1024; i += 256) {
      int r = i >> 5, c4 = i & 31;
      float4 d = *(const float4*)(W + (size_t)(kt + r) * D_EMB + n0 + c4 * 4);
      int n = c4 * 4;
      Bt[(n + 0) * 40 + r] = f2bf(d.x);
      Bt[(n + 1) * 40 + r] = f2bf(d.y);
      Bt[(n + 2) * 40 + r] = f2bf(d.z);
      Bt[(n + 3) * 40 + r] = f2bf(d.w);
    }
    __syncthreads();

    v16bf af[2], bfrag[4];
    for (int mi = 0; mi < 2; ++mi)
      af[mi] = load_a_frag(At, wm * 32 + mi * 16 + (lane & 15), 40, 0, lane);
    for (int ni = 0; ni < 4; ++ni)
      bfrag[ni] = load_b_frag(Bt, wn * 64 + ni * 16 + (lane & 15), 40, 0, lane);
    for (int mi = 0; mi < 2; ++mi)
      for (int ni = 0; ni < 4; ++ni)
        acc[mi][ni] = __builtin_amdgcn_wmma_f32_16x16x32_bf16(
            false, af[mi], false, bfrag[ni], (short)0, acc[mi][ni], false, false);
    __syncthreads();
  }

  // epilogue: + bias, bf16, scatter into [B, H, T, Hd]
  for (int mi = 0; mi < 2; ++mi)
    for (int ni = 0; ni < 4; ++ni) {
      int n = n0 + wn * 64 + ni * 16 + (lane & 15);
      float bb = bias[n];
      int h = n >> 6, hd = n & 63;
      for (int v = 0; v < 8; ++v) {
        int m = m0 + wm * 32 + mi * 16 + v + ((lane & 16) ? 8 : 0);
        int bidx = m >> 11, t = m & 2047;
        out[((size_t)(bidx * N_HEADS + h) * T_SEQ + t) * HD + hd] =
            f2bf(acc[mi][ni][v] + bb);
      }
    }
}

// ---------------------------------------------------------------------------
// Kernel 2: flash attention. grid = (T/128, H, B), block = 256 (8 waves).
// Each wave owns 16 query rows; online softmax over 64-key tiles.
// ---------------------------------------------------------------------------
__global__ __launch_bounds__(256) void attn_kernel(
    const unsigned short* __restrict__ Qb, const unsigned short* __restrict__ Kb,
    const unsigned short* __restrict__ Vb, unsigned short* __restrict__ ctx)
{
  __shared__ unsigned short Kt[64 * 72];       // K tile [key][hd]
  __shared__ unsigned short Vt[64 * 72];       // V tile transposed [hd][key]
  __shared__ unsigned short Pb[8 * 16 * 72];   // per-wave P scratch

  int tid = threadIdx.x, lane = tid & 31, wave = tid >> 5;
  int b = blockIdx.z, h = blockIdx.y;
  int qbase = blockIdx.x * 128 + wave * 16;

  const unsigned short* Qh = Qb + (size_t)(b * N_HEADS + h) * T_SEQ * HD;
  const unsigned short* Kh = Kb + (size_t)(b * N_HEADS + h) * T_SEQ * HD;
  const unsigned short* Vh = Vb + (size_t)(b * N_HEADS + h) * T_SEQ * HD;

  // Q A-fragments (16 x 64, two K=32 chunks) straight from global
  v16bf aq[2];
  {
    const unsigned short* qp = Qh + (size_t)(qbase + (lane & 15)) * HD;
    for (int c = 0; c < 2; ++c) {
      FragU u; int k0 = 32 * c + ((lane & 16) ? 8 : 0);
      u.q[0] = *(const uint4*)(qp + k0);
      u.q[1] = *(const uint4*)(qp + k0 + 16);
      aq[c] = u.v;
    }
  }

  float mrow[8], lrow[8];
  v8f o[4];
  for (int v = 0; v < 8; ++v) { mrow[v] = -1e30f; lrow[v] = 0.f; }
  for (int f = 0; f < 4; ++f)
    for (int v = 0; v < 8; ++v) o[f][v] = 0.f;

  unsigned short* P = &Pb[wave * 16 * 72];
  int rbase = (lane & 16) ? 8 : 0;

  for (int kt = 0; kt < T_SEQ; kt += 64) {
    // stage K tile (natural layout == N-major B layout for Q*K^T)
#if defined(HAVE_ASYNC_LDS)
    for (int i = tid; i < 512; i += 256) {
      int key = i >> 3, c8 = i & 7;
      __builtin_amdgcn_global_load_async_to_lds_b128(
          (v4i*)(Kh + (size_t)(kt + key) * HD + c8 * 8),
          (v4i*)(&Kt[key * 72 + c8 * 8]), 0, 0);
    }
#else
    for (int i = tid; i < 512; i += 256) {
      int key = i >> 3, c8 = i & 7;
      uint4 d = *(const uint4*)(Kh + (size_t)(kt + key) * HD + c8 * 8);
      *(uint4*)(&Kt[key * 72 + c8 * 8]) = d;
    }
#endif
    // stage V tile transposed (N-major B layout for P*V)
    for (int i = tid; i < 512; i += 256) {
      int key = i >> 3, c8 = i & 7;
      uint4 d = *(const uint4*)(Vh + (size_t)(kt + key) * HD + c8 * 8);
      const unsigned short* s = (const unsigned short*)&d;
#pragma unroll
      for (int j = 0; j < 8; ++j) Vt[(c8 * 8 + j) * 72 + key] = s[j];
    }
    wait_async_cnt0();
    __syncthreads();

    // S = (Q K^T) / sqrt(Hd)  — 4 n-frags x 2 K-chunks
    v8f sc[4];
    for (int f = 0; f < 4; ++f) {
      for (int v = 0; v < 8; ++v) sc[f][v] = 0.f;
      for (int c = 0; c < 2; ++c) {
        v16bf bk = load_b_frag(Kt, f * 16 + (lane & 15), 72, 32 * c, lane);
        sc[f] = __builtin_amdgcn_wmma_f32_16x16x32_bf16(
            false, aq[c], false, bk, (short)0, sc[f], false, false);
      }
      for (int v = 0; v < 8; ++v) sc[f][v] = sc[f][v] * 0.125f;
    }

    // online softmax; C-frag row v lives in 16-lane half -> xor masks 1..8
    for (int v = 0; v < 8; ++v) {
      float m = sc[0][v];
      m = fmaxf(m, sc[1][v]); m = fmaxf(m, sc[2][v]); m = fmaxf(m, sc[3][v]);
      for (int msk = 1; msk < 16; msk <<= 1) m = fmaxf(m, __shfl_xor(m, msk, 32));
      float mn = fmaxf(mrow[v], m);
      float fac = __expf(mrow[v] - mn);
      mrow[v] = mn;
      float rs = 0.f;
      for (int f = 0; f < 4; ++f) {
        float p = __expf(sc[f][v] - mn);
        sc[f][v] = p; rs += p;
      }
      for (int msk = 1; msk < 16; msk <<= 1) rs += __shfl_xor(rs, msk, 32);
      lrow[v] = lrow[v] * fac + rs;
      for (int f = 0; f < 4; ++f) o[f][v] = o[f][v] * fac;
    }

    // dump P to wave-private LDS, re-read in A-fragment layout (DS in-order/wave)
    for (int f = 0; f < 4; ++f)
      for (int v = 0; v < 8; ++v)
        P[(v + rbase) * 72 + f * 16 + (lane & 15)] = f2bf(sc[f][v]);

    v16bf ap[2];
    for (int c = 0; c < 2; ++c)
      ap[c] = load_a_frag(P, lane & 15, 72, 32 * c, lane);

    for (int fn = 0; fn < 4; ++fn)
      for (int c = 0; c < 2; ++c) {
        v16bf bv2 = load_b_frag(Vt, fn * 16 + (lane & 15), 72, 32 * c, lane);
        o[fn] = __builtin_amdgcn_wmma_f32_16x16x32_bf16(
            false, ap[c], false, bv2, (short)0, o[fn], false, false);
      }
    __syncthreads();
  }

  // normalize and write ctx as bf16 [B, T, D] (row-major for the out-proj GEMM)
  for (int fn = 0; fn < 4; ++fn)
    for (int v = 0; v < 8; ++v) {
      int t = qbase + v + rbase;
      int hd = fn * 16 + (lane & 15);
      ctx[(size_t)(b * T_SEQ + t) * D_EMB + h * HD + hd] = f2bf(o[fn][v] / lrow[v]);
    }
}

// ---------------------------------------------------------------------------
// Kernel 3: output projection + residual: y = ctx @ Wo + bo + x   (fp32 out)
// Block tile 128x128, wave tile 32x64.
// ---------------------------------------------------------------------------
__global__ __launch_bounds__(256) void proj_kernel(
    const unsigned short* __restrict__ ctx, const float* __restrict__ wo,
    const float* __restrict__ bo, const float* __restrict__ x,
    float* __restrict__ y)
{
  __shared__ unsigned short At[128 * 40];
  __shared__ unsigned short Bt[128 * 40];

  int tid = threadIdx.x, lane = tid & 31, wave = tid >> 5;
  int wm = wave >> 1, wn = wave & 1;
  int m0 = blockIdx.y * 128, n0 = blockIdx.x * 128;

  v8f acc[2][4];
  for (int mi = 0; mi < 2; ++mi)
    for (int ni = 0; ni < 4; ++ni)
      for (int v = 0; v < 8; ++v) acc[mi][ni][v] = 0.f;

  for (int kt = 0; kt < D_EMB; kt += 32) {
    for (int i = tid; i < 512; i += 256) {   // A already bf16: 512 uint4 slots
      int r = i >> 2, c8 = i & 3;
      uint4 d = *(const uint4*)(ctx + (size_t)(m0 + r) * D_EMB + kt + c8 * 8);
      *(uint4*)(&At[r * 40 + c8 * 8]) = d;
    }
    for (int i = tid; i < 1024; i += 256) {  // Wo: 32x128 fp32 -> Bt[n][k]
      int r = i >> 5, c4 = i & 31;
      float4 d = *(const float4*)(wo + (size_t)(kt + r) * D_EMB + n0 + c4 * 4);
      int n = c4 * 4;
      Bt[(n + 0) * 40 + r] = f2bf(d.x);
      Bt[(n + 1) * 40 + r] = f2bf(d.y);
      Bt[(n + 2) * 40 + r] = f2bf(d.z);
      Bt[(n + 3) * 40 + r] = f2bf(d.w);
    }
    __syncthreads();

    v16bf af[2], bfrag[4];
    for (int mi = 0; mi < 2; ++mi)
      af[mi] = load_a_frag(At, wm * 32 + mi * 16 + (lane & 15), 40, 0, lane);
    for (int ni = 0; ni < 4; ++ni)
      bfrag[ni] = load_b_frag(Bt, wn * 64 + ni * 16 + (lane & 15), 40, 0, lane);
    for (int mi = 0; mi < 2; ++mi)
      for (int ni = 0; ni < 4; ++ni)
        acc[mi][ni] = __builtin_amdgcn_wmma_f32_16x16x32_bf16(
            false, af[mi], false, bfrag[ni], (short)0, acc[mi][ni], false, false);
    __syncthreads();
  }

  for (int mi = 0; mi < 2; ++mi)
    for (int ni = 0; ni < 4; ++ni) {
      int n = n0 + wn * 64 + ni * 16 + (lane & 15);
      float bb = bo[n];
      for (int v = 0; v < 8; ++v) {
        int m = m0 + wm * 32 + mi * 16 + v + ((lane & 16) ? 8 : 0);
        size_t idx = (size_t)m * D_EMB + n;
        y[idx] = acc[mi][ni][v] + bb + x[idx];
      }
    }
}

// ---------------------------------------------------------------------------
// Kernel 4: LayerNorm per 1024-wide row. grid = 4096, block = 256.
// ---------------------------------------------------------------------------
__global__ __launch_bounds__(256) void ln_kernel(
    const float* __restrict__ y, const float* __restrict__ gamma,
    const float* __restrict__ beta, float* __restrict__ out)
{
  __shared__ float r1[8], r2[8];
  int row = blockIdx.x, tid = threadIdx.x, lane = tid & 31, wave = tid >> 5;
  const float* yr = y + (size_t)row * D_EMB;
  float4 d = *(const float4*)(yr + tid * 4);
  float s1 = d.x + d.y + d.z + d.w;
  float s2 = d.x * d.x + d.y * d.y + d.z * d.z + d.w * d.w;
  for (int m = 1; m < 32; m <<= 1) {
    s1 += __shfl_xor(s1, m, 32);
    s2 += __shfl_xor(s2, m, 32);
  }
  if (lane == 0) { r1[wave] = s1; r2[wave] = s2; }
  __syncthreads();
  if (tid == 0) {
    float a = 0.f, b2 = 0.f;
    for (int i = 0; i < 8; ++i) { a += r1[i]; b2 += r2[i]; }
    r1[0] = a; r2[0] = b2;
  }
  __syncthreads();
  float mu  = r1[0] * (1.0f / 1024.0f);
  float var = r2[0] * (1.0f / 1024.0f) - mu * mu;
  float inv = rsqrtf(var + 1e-5f);
  int c = tid * 4;
  float4 g = *(const float4*)(gamma + c);
  float4 bt = *(const float4*)(beta + c);
  float4 o;
  o.x = (d.x - mu) * inv * g.x + bt.x;
  o.y = (d.y - mu) * inv * g.y + bt.y;
  o.z = (d.z - mu) * inv * g.z + bt.z;
  o.w = (d.w - mu) * inv * g.w + bt.w;
  *(float4*)(out + (size_t)row * D_EMB + c) = o;
}

extern "C" void kernel_launch(void* const* d_in, const int* in_sizes, int n_in,
                              void* d_out, int out_size, void* d_ws, size_t ws_size,
                              hipStream_t stream) {
  (void)in_sizes; (void)n_in; (void)out_size; (void)ws_size;
  const float* x     = (const float*)d_in[0];
  const float* wq    = (const float*)d_in[1];
  const float* bq    = (const float*)d_in[2];
  const float* wk    = (const float*)d_in[3];
  const float* bk    = (const float*)d_in[4];
  const float* wv    = (const float*)d_in[5];
  const float* bv    = (const float*)d_in[6];
  const float* wo    = (const float*)d_in[7];
  const float* bo    = (const float*)d_in[8];
  const float* gamma = (const float*)d_in[9];
  const float* beta  = (const float*)d_in[10];
  float* out = (float*)d_out;

  char* ws = (char*)d_ws;
  const size_t elems = (size_t)2 * T_SEQ * D_EMB;          // 4,194,304
  unsigned short* Qb  = (unsigned short*)(ws);             // bf16 [B,H,T,Hd]
  unsigned short* Kb  = (unsigned short*)(ws + elems * 2);
  unsigned short* Vb  = (unsigned short*)(ws + elems * 4);
  unsigned short* ctx = (unsigned short*)(ws + elems * 6); // bf16 [B,T,D]
  float*          yb  = (float*)(ws + elems * 8);          // fp32 [B,T,D]

  qkv_gemm_kernel<<<dim3(8, 32, 3), 256, 0, stream>>>(x, wq, bq, wk, bk, wv, bv,
                                                      Qb, Kb, Vb);
  attn_kernel<<<dim3(16, N_HEADS, 2), 256, 0, stream>>>(Qb, Kb, Vb, ctx);
  proj_kernel<<<dim3(8, 32), 256, 0, stream>>>(ctx, wo, bo, x, yb);
  ln_kernel<<<dim3(4096), 256, 0, stream>>>(yb, gamma, beta, out);
}
